// DS_MultiHeadLatentAttention_80212809220546
// MI455X (gfx1250) — matrix operand order repaired and checked by reading, not compiled, not run on hardware
//
#include <hip/hip_runtime.h>
#include <hip/hip_bf16.h>

typedef _Float16 f16x8 __attribute__((ext_vector_type(8)));
typedef _Float16 v16h  __attribute__((ext_vector_type(16)));
typedef float    v8f   __attribute__((ext_vector_type(8)));

#define WMMA_F16(A,B,C) __builtin_amdgcn_wmma_f32_16x16x32_f16(false,(A),false,(B),(short)0,(C),false,false)

// A-operand fragment (16x32 f16): row = lane&15, K contiguous 16 values at [hv*16, +16)
__device__ __forceinline__ v16h load_afrag(const _Float16* p) {
    union { v16h v; f16x8 h[2]; } u;
    u.h[0] = *(const f16x8*)(p);
    u.h[1] = *(const f16x8*)(p + 8);
    return u.v;
}
// B-operand fragment (32x16 f16, supplied as NxK row-major): col = lane&15,
// K contiguous 16 values at [hv*16, +16)
__device__ __forceinline__ v16h load_bfrag(const _Float16* p) {
    union { v16h v; f16x8 h[2]; } u;
    u.h[0] = *(const f16x8*)(p);
    u.h[1] = *(const f16x8*)(p + 8);
    return u.v;
}

// gfx1250 async copy: global -> LDS, 16 bytes per lane, tracked by ASYNCcnt.
__device__ __forceinline__ void async_copy_b128(const _Float16* gsrc, _Float16* ldst) {
    unsigned lds_addr = (unsigned)(uintptr_t)ldst; // low 32 bits of generic shared ptr = LDS offset
    asm volatile("global_load_async_to_lds_b128 %0, %1, off"
                 :: "v"(lds_addr), "v"(gsrc) : "memory");
}

// ---------------- elementwise conversions ----------------
__global__ void cvt_f32_f16(const float* __restrict__ src, _Float16* __restrict__ dst, int n) {
    int i = blockIdx.x * blockDim.x + threadIdx.x;
    if (i < n) dst[i] = (_Float16)src[i];
}

// W [rows x cols] f32 -> Wt [cols x rows] f16  (N x K for WMMA B-operand)
__global__ void cvt_transpose_f16(const float* __restrict__ W, _Float16* __restrict__ Wt,
                                  int rows, int cols) {
    int i = blockIdx.x * blockDim.x + threadIdx.x;
    if (i < rows * cols) {
        int n = i / rows;
        int k = i - n * rows;
        Wt[i] = (_Float16)W[(size_t)k * cols + n];
    }
}

// ---------------- WMMA GEMM with async-LDS double-buffered staging ----------------
// C[M x N] = A[M x K](f16) * Bt[N x K](f16)^T + bias. 256 threads = 8 waves,
// block tile 128x64, 32x32 per wave, K-step 32. A/B K-slabs staged in LDS via
// GLOBAL_LOAD_ASYNC_TO_LDS_B128 (3 per thread per step), double-buffered.
// MODE 0: f16 row-major. MODE 1: f16 scattered [b,h,dk,s] (V^T). MODE 2: f32.
template<int MODE>
__global__ __launch_bounds__(256)
void gemm_wmma_kernel(const _Float16* __restrict__ A, const _Float16* __restrict__ Bt,
                      const float* __restrict__ bias, void* __restrict__ Cout,
                      int M, int N, int K) {
    __shared__ __align__(16) _Float16 sA[2][128 * 32]; // [row][k]
    __shared__ __align__(16) _Float16 sB[2][64 * 32];  // [col][k]

    int tid  = threadIdx.x;
    int lane = tid & 31;
    int w    = tid >> 5;
    int wm   = w >> 1, wn = w & 1;
    int rowbase = blockIdx.y * 128;
    int colbase = blockIdx.x * 64;
    int r  = lane & 15;
    int hv = lane >> 4;

    // per-thread copy assignment for one K-slab
    int arow = tid >> 1;            // 0..127, two 8-f16 chunks each
    int acol = (tid & 1) * 16;      // 0 or 16 (f16 units)
    int brow = tid >> 2;            // 0..63, one 8-f16 chunk each
    int bcol = (tid & 3) * 8;       // 0,8,16,24

    const _Float16* gA = A  + (size_t)(rowbase + arow) * K + acol;
    const _Float16* gB = Bt + (size_t)(colbase + brow) * K + bcol;
    _Float16* lA = &sA[0][arow * 32 + acol];
    _Float16* lB = &sB[0][brow * 32 + bcol];
    const int bufstep_a = 128 * 32, bufstep_b = 64 * 32;

    // preload slab 0 into buffer 0
    async_copy_b128(gA,     lA);
    async_copy_b128(gA + 8, lA + 8);
    async_copy_b128(gB,     lB);

    v8f c00 = {}, c01 = {}, c10 = {}, c11 = {};
    int nsteps = K >> 5;
    for (int i = 0; i < nsteps; ++i) {
        int cur = i & 1;
        if (i + 1 < nsteps) {
            int kk = (i + 1) << 5;
            int nxt = cur ^ 1;
            async_copy_b128(gA + kk,     lA + nxt * bufstep_a);
            async_copy_b128(gA + kk + 8, lA + nxt * bufstep_a + 8);
            async_copy_b128(gB + kk,     lB + nxt * bufstep_b);
            asm volatile("s_wait_asynccnt 0x3" ::: "memory"); // slab i landed
        } else {
            asm volatile("s_wait_asynccnt 0x0" ::: "memory");
        }
        __syncthreads();

        const _Float16* pa0 = &sA[cur][(wm * 32 + r) * 32 + hv * 16];
        const _Float16* pb0 = &sB[cur][(wn * 32 + r) * 32 + hv * 16];
        v16h a0 = load_afrag(pa0);
        v16h a1 = load_afrag(pa0 + 16 * 32);
        v16h b0 = load_bfrag(pb0);
        v16h b1 = load_bfrag(pb0 + 16 * 32);
        c00 = WMMA_F16(a0, b0, c00);
        c01 = WMMA_F16(a0, b1, c01);
        c10 = WMMA_F16(a1, b0, c10);
        c11 = WMMA_F16(a1, b1, c11);
        __syncthreads(); // buffer cur is free to be overwritten next step
    }

    v8f* acc[2][2] = { {&c00, &c01}, {&c10, &c11} };
    #pragma unroll
    for (int i = 0; i < 2; ++i) {
        #pragma unroll
        for (int j = 0; j < 2; ++j) {
            int ncol = colbase + wn * 32 + j * 16 + r;
            float bv = bias[ncol];
            #pragma unroll
            for (int vi = 0; vi < 8; ++vi) {
                int m = rowbase + wm * 32 + i * 16 + vi + 8 * hv;
                float val = (*acc[i][j])[vi] + bv;
                if (MODE == 0) {
                    ((_Float16*)Cout)[(size_t)m * N + ncol] = (_Float16)val;
                } else if (MODE == 2) {
                    ((float*)Cout)[(size_t)m * N + ncol] = val;
                } else {
                    // m = b*2048 + s ; ncol = h*64 + dk -> Vt[((b*16+h)*64+dk)*2048 + s]
                    int b  = m >> 11, s = m & 2047;
                    int h  = ncol >> 6, dk = ncol & 63;
                    ((_Float16*)Cout)[(((size_t)(b * 16 + h) * 64 + dk) << 11) + s] = (_Float16)val;
                }
            }
        }
    }
}

// ---------------- flash attention, one wave per (b, h, 16-query block) ----------------
__global__ __launch_bounds__(32)
void mla_attn_kernel(const _Float16* __restrict__ Q,   // [b*L, 1024] (l, h*64+dk)
                     const _Float16* __restrict__ Km,  // [b*S, 1024]
                     const _Float16* __restrict__ Vt,  // [b, h, 64, S]
                     _Float16* __restrict__ Oout) {    // [b*L, 1024]
    constexpr int L = 2048, H = 16, DM = 1024, S = 2048;
    const float scale = 0.125f; // 1/sqrt(64)
    int qb = blockIdx.x, h = blockIdx.y, b = blockIdx.z;
    int lane = threadIdx.x & 31;
    int r = lane & 15, hv = lane >> 4;

    __shared__ __align__(16) _Float16 pt[16 * 32];

    size_t qrow = (size_t)b * L + qb * 16;
    const _Float16* qbase = Q + (qrow + r) * DM + h * 64 + hv * 16;
    v16h qa0 = load_afrag(qbase);
    v16h qa1 = load_afrag(qbase + 32);

    v8f o0 = {}, o1 = {}, o2 = {}, o3 = {};
    float rmax[8], rsum[8];
    #pragma unroll
    for (int vi = 0; vi < 8; ++vi) { rmax[vi] = -1e30f; rsum[vi] = 0.f; }

    const _Float16* kbb = Km + ((size_t)b * S) * DM + h * 64 + hv * 16;
    const _Float16* vbb = Vt + ((size_t)b * H + h) * 64 * S;
    int nch = (qb * 16 + 16 + 31) >> 5;

    for (int sc = 0; sc < nch; ++sc) {
        int sbase = sc * 32;
        v8f st0 = {}, st1 = {};
        {
            v16h kb;
            kb  = load_bfrag(kbb + (size_t)(sbase + r) * DM);
            st0 = WMMA_F16(qa0, kb, st0);
            kb  = load_bfrag(kbb + (size_t)(sbase + r) * DM + 32);
            st0 = WMMA_F16(qa1, kb, st0);
            kb  = load_bfrag(kbb + (size_t)(sbase + 16 + r) * DM);
            st1 = WMMA_F16(qa0, kb, st1);
            kb  = load_bfrag(kbb + (size_t)(sbase + 16 + r) * DM + 32);
            st1 = WMMA_F16(qa1, kb, st1);
        }
        float tmax[8], tsum[8], alpha[8];
        #pragma unroll
        for (int vi = 0; vi < 8; ++vi) {
            int qi = qb * 16 + vi + 8 * hv;          // query row of this element
            float v0 = st0[vi] * scale; if (sbase + r      > qi) v0 = -1e30f;
            float v1 = st1[vi] * scale; if (sbase + 16 + r > qi) v1 = -1e30f;
            st0[vi] = v0; st1[vi] = v1;
            tmax[vi] = fmaxf(v0, v1);
        }
        #pragma unroll
        for (int vi = 0; vi < 8; ++vi) {
            #pragma unroll
            for (int off = 1; off < 16; off <<= 1)
                tmax[vi] = fmaxf(tmax[vi], __shfl_xor(tmax[vi], off, 32));
            float nm = fmaxf(rmax[vi], tmax[vi]);
            alpha[vi] = __expf(rmax[vi] - nm);
            rmax[vi]  = nm;
        }
        #pragma unroll
        for (int vi = 0; vi < 8; ++vi) {
            float p0 = __expf(st0[vi] - rmax[vi]);
            float p1 = __expf(st1[vi] - rmax[vi]);
            st0[vi] = p0; st1[vi] = p1;
            tsum[vi] = p0 + p1;
        }
        #pragma unroll
        for (int vi = 0; vi < 8; ++vi) {
            #pragma unroll
            for (int off = 1; off < 16; off <<= 1)
                tsum[vi] += __shfl_xor(tsum[vi], off, 32);
            rsum[vi] = rsum[vi] * alpha[vi] + tsum[vi];
            o0[vi] *= alpha[vi]; o1[vi] *= alpha[vi];
            o2[vi] *= alpha[vi]; o3[vi] *= alpha[vi];
        }
        // P tile (f32 D-layout) -> LDS f16 -> A-fragment layout
        #pragma unroll
        for (int vi = 0; vi < 8; ++vi) {
            pt[(vi + 8 * hv) * 32 + r]      = (_Float16)st0[vi];
            pt[(vi + 8 * hv) * 32 + 16 + r] = (_Float16)st1[vi];
        }
        __syncthreads();
        v16h pa = load_afrag(pt + r * 32 + hv * 16);
        v16h vb;
        vb = load_bfrag(vbb + (size_t)(0 * 16 + r) * S + sbase + hv * 16);
        o0 = WMMA_F16(pa, vb, o0);
        vb = load_bfrag(vbb + (size_t)(1 * 16 + r) * S + sbase + hv * 16);
        o1 = WMMA_F16(pa, vb, o1);
        vb = load_bfrag(vbb + (size_t)(2 * 16 + r) * S + sbase + hv * 16);
        o2 = WMMA_F16(pa, vb, o2);
        vb = load_bfrag(vbb + (size_t)(3 * 16 + r) * S + sbase + hv * 16);
        o3 = WMMA_F16(pa, vb, o3);
        __syncthreads();
    }

    #pragma unroll
    for (int vi = 0; vi < 8; ++vi) {
        float inv = 1.0f / rsum[vi];
        int m = vi + 8 * hv;
        size_t orow = (qrow + m) * DM + h * 64;
        Oout[orow + 0 * 16 + r] = (_Float16)(o0[vi] * inv);
        Oout[orow + 1 * 16 + r] = (_Float16)(o1[vi] * inv);
        Oout[orow + 2 * 16 + r] = (_Float16)(o2[vi] * inv);
        Oout[orow + 3 * 16 + r] = (_Float16)(o3[vi] * inv);
    }
}

extern "C" void kernel_launch(void* const* d_in, const int* in_sizes, int n_in,
                              void* d_out, int out_size, void* d_ws, size_t ws_size,
                              hipStream_t stream) {
    constexpr int B = 2, L = 2048, DM = 1024, LAT = 256, H = 16;
    const int M = B * L; // 4096

    const float* queries = (const float*)d_in[0];
    const float* keys    = (const float*)d_in[1];
    // d_in[2] (values) is unused by the reference
    const float* Wq = (const float*)d_in[3];  const float* bq = (const float*)d_in[4];
    const float* Wd = (const float*)d_in[5];  const float* bd = (const float*)d_in[6];
    const float* Wk = (const float*)d_in[7];  const float* bk = (const float*)d_in[8];
    const float* Wv = (const float*)d_in[9];  const float* bv = (const float*)d_in[10];
    const float* Wo = (const float*)d_in[11]; const float* bo = (const float*)d_in[12];
    float* out = (float*)d_out;

    const size_t E = (size_t)M * DM; // 4 Mi elements
    _Float16* qf   = (_Float16*)d_ws;
    _Float16* kf   = qf   + E;
    _Float16* WqT  = kf   + E;                         // [1024 x 1024]
    _Float16* WdT  = WqT  + (size_t)DM * DM;           // [256 x 1024]
    _Float16* WkT  = WdT  + (size_t)LAT * DM;          // [1024 x 256]
    _Float16* WvT  = WkT  + (size_t)DM * LAT;          // [1024 x 256]
    _Float16* WoT  = WvT  + (size_t)DM * LAT;          // [1024 x 1024]
    _Float16* Qp   = WoT  + (size_t)DM * DM;           // [4096 x 1024]
    _Float16* lat  = Qp   + E;                         // [4096 x 256]
    _Float16* Kp   = lat  + (size_t)M * LAT;           // [4096 x 1024]
    _Float16* Vt   = Kp   + E;                         // [b,h,64,2048]
    _Float16* attn = Vt   + E;                         // [4096 x 1024]

    // convert activations
    cvt_f32_f16<<<(int)(E / 256), 256, 0, stream>>>(queries, qf, (int)E);
    cvt_f32_f16<<<(int)(E / 256), 256, 0, stream>>>(keys,    kf, (int)E);
    // convert + transpose weights to NxK
    cvt_transpose_f16<<<(DM * DM) / 256, 256, 0, stream>>>(Wq, WqT, DM, DM);
    cvt_transpose_f16<<<(DM * LAT) / 256, 256, 0, stream>>>(Wd, WdT, DM, LAT);
    cvt_transpose_f16<<<(LAT * DM) / 256, 256, 0, stream>>>(Wk, WkT, LAT, DM);
    cvt_transpose_f16<<<(LAT * DM) / 256, 256, 0, stream>>>(Wv, WvT, LAT, DM);
    cvt_transpose_f16<<<(DM * DM) / 256, 256, 0, stream>>>(Wo, WoT, DM, DM);

    dim3 blk(256);
    // Q = q @ Wq + bq
    gemm_wmma_kernel<0><<<dim3(DM / 64, M / 128), blk, 0, stream>>>(qf, WqT, bq, Qp, M, DM, DM);
    // latent = k @ Wd + bd
    gemm_wmma_kernel<0><<<dim3(LAT / 64, M / 128), blk, 0, stream>>>(kf, WdT, bd, lat, M, LAT, DM);
    // K = latent @ Wk + bk
    gemm_wmma_kernel<0><<<dim3(DM / 64, M / 128), blk, 0, stream>>>(lat, WkT, bk, Kp, M, DM, LAT);
    // V^T = (latent @ Wv + bv) stored [b,h,dk,s]
    gemm_wmma_kernel<1><<<dim3(DM / 64, M / 128), blk, 0, stream>>>(lat, WvT, bv, Vt, M, DM, LAT);
    // attention
    mla_attn_kernel<<<dim3(L / 16, H, B), 32, 0, stream>>>(Qp, Kp, Vt, attn);
    // out = attn @ Wo + bo (f32)
    gemm_wmma_kernel<2><<<dim3(DM / 64, M / 128), blk, 0, stream>>>(attn, WoT, bo, out, M, DM, DM);
}
